// GraphSAGE_16381005267298
// MI455X (gfx1250) — compile-verified, matrix operand order checked by reading, not run on hardware
//
#include <hip/hip_runtime.h>

#define N_NODES 100000
#define N_EDGES 640000
#define DDIM    128

typedef __attribute__((ext_vector_type(16))) __bf16 v16bf;
typedef __attribute__((ext_vector_type(8)))  __bf16 v8bf;
typedef __attribute__((ext_vector_type(4)))  __bf16 v4bf;
typedef __attribute__((ext_vector_type(8)))  float  v8f;

static __device__ inline v16bf combine16(v8bf lo, v8bf hi) {
    v16bf r;
#pragma unroll
    for (int i = 0; i < 8; ++i) { r[i] = lo[i]; r[i + 8] = hi[i]; }
    return r;
}

// ---------------------------------------------------------------- utilities
__global__ void zero_f32_kernel(float* p, int n4) {
    int i = blockIdx.x * blockDim.x + threadIdx.x;
    if (i < n4) ((float4*)p)[i] = make_float4(0.f, 0.f, 0.f, 0.f);
}

__global__ void cvt_f32_to_bf16_kernel(const float* __restrict__ in,
                                       __bf16* __restrict__ out, int n) {
    int i = (blockIdx.x * blockDim.x + threadIdx.x) * 4;
    if (i + 3 < n) {
        float4 v = *(const float4*)(in + i);
        v4bf o;
        o[0] = (__bf16)v.x; o[1] = (__bf16)v.y;
        o[2] = (__bf16)v.z; o[3] = (__bf16)v.w;
        *(v4bf*)(out + i) = o;
    }
}

// Pack combined [W_l ; W_r] (K=256) into WMMA-B fragment-major bf16 layout.
// Fragment (ks, ct): 32 lanes x 16 elems, per ISA 16-bit B layout:
//   lane<16 : n = ct*16+lane,    k = kb + (e<8 ? e   : e+8)
//   lane>=16: n = ct*16+lane-16, k = kb + (e<8 ? e+8 : e+16)
__global__ void prep_wfrag_kernel(const float* __restrict__ Wl,
                                  const float* __restrict__ Wr,
                                  __bf16* __restrict__ frag) {
    int t = blockIdx.x * blockDim.x + threadIdx.x;   // 2048 threads
    if (t >= 8 * 8 * 32) return;
    int lane = t & 31;
    int ct   = (t >> 5) & 7;
    int ks   = (t >> 8) & 7;
    int kb   = ks * 32;
    int n    = ct * 16 + (lane & 15);
    __bf16* dst = frag + ((size_t)(ks * 8 + ct) * 32 + lane) * 16;
#pragma unroll
    for (int e = 0; e < 16; ++e) {
        int k = (lane < 16) ? (kb + (e < 8 ? e : e + 8))
                            : (kb + (e < 8 ? e + 8 : e + 16));
        float v = (k < DDIM) ? Wl[n * DDIM + k] : Wr[n * DDIM + (k - DDIM)];
        dst[e] = (__bf16)v;
    }
}

// ------------------------------------------------------------- aggregation
// One edge per 32 lanes; each lane moves 4 dims (8B bf16 load, 4 f32 atomics).
__global__ void aggregate_kernel(const __bf16* __restrict__ h,
                                 const long long* __restrict__ src,
                                 const long long* __restrict__ dst,
                                 float* __restrict__ agg,
                                 float* __restrict__ cnt,
                                 int nE, int withCnt) {
    int e    = blockIdx.x * 8 + (threadIdx.x >> 5);
    int lane = threadIdx.x & 31;
    if (e >= nE) return;
    int s = (int)src[e];
    int d = (int)dst[e];
    v4bf m = *(const v4bf*)(h + (size_t)s * DDIM + lane * 4);
    float* ap = agg + (size_t)d * DDIM + lane * 4;
    atomicAdd(ap + 0, (float)m[0]);
    atomicAdd(ap + 1, (float)m[1]);
    atomicAdd(ap + 2, (float)m[2]);
    atomicAdd(ap + 3, (float)m[3]);
    if (withCnt && lane == 0) atomicAdd(cnt + d, 1.0f);
}

__global__ void finalize_mean_kernel(const float* __restrict__ agg,
                                     const float* __restrict__ cnt,
                                     __bf16* __restrict__ mean) {
    int base = (blockIdx.x * blockDim.x + threadIdx.x) * 4;
    if (base >= N_NODES * DDIM) return;
    int row = base >> 7;
    float inv = 1.0f / fmaxf(cnt[row], 1.0f);
    float4 v = *(const float4*)(agg + base);
    v4bf o;
    o[0] = (__bf16)(v.x * inv); o[1] = (__bf16)(v.y * inv);
    o[2] = (__bf16)(v.z * inv); o[3] = (__bf16)(v.w * inv);
    *(v4bf*)(mean + base) = o;
}

// -------------------------------------------------------------- WMMA GEMM
// out[rb:rb+16, 0:128] = [mean | self] @ Wc^T + bias  (one wave per 16-row strip)
// K = 256 (k<128 -> mean, k>=128 -> self), 8 K-steps x 8 column tiles = 64 WMMA.
__global__ void __launch_bounds__(256)
sage_gemm_kernel(const __bf16* __restrict__ meanB,
                 const __bf16* __restrict__ selfB,
                 const __bf16* __restrict__ wfrag,   // 64KB fragment-major
                 const float*  __restrict__ bias,
                 float*  __restrict__ outF,
                 __bf16* __restrict__ outB,
                 int nTiles, int doRelu) {
    __shared__ __bf16 sW[32768];                     // 64 KB
    int tid = threadIdx.x;

    // cooperative 64KB global -> LDS stage (b128 coalesced)
    {
        const uint4* gs = (const uint4*)wfrag;
        uint4* ls = (uint4*)sW;
        for (int i = tid; i < 4096; i += 256) ls[i] = gs[i];
    }
    __syncthreads();

    int wave = tid >> 5;
    int lane = tid & 31;
    int tile = blockIdx.x * 8 + wave;
    if (tile >= nTiles) return;
    int rb    = tile * 16;
    int lhalf = lane >> 4;       // 0 | 1
    int l15   = lane & 15;

    v8f acc[8];
#pragma unroll
    for (int ct = 0; ct < 8; ++ct)
#pragma unroll
        for (int r = 0; r < 8; ++r) acc[ct][r] = 0.0f;

#pragma unroll
    for (int ks = 0; ks < 8; ++ks) {
        int kb = ks * 32;
        const __bf16* asrc = (kb < DDIM) ? meanB : selfB;
        int k0 = (kb < DDIM) ? kb : (kb - DDIM);
        // A fragment: lane<16 -> K {k0..k0+7, k0+16..23}; lane>=16 -> +8
        int row = rb + l15;
        const __bf16* ap = asrc + (size_t)row * DDIM + k0 + lhalf * 8;
        v8bf alo = *(const v8bf*)ap;
        v8bf ahi = *(const v8bf*)(ap + 16);
        v16bf a = combine16(alo, ahi);
#pragma unroll
        for (int ct = 0; ct < 8; ++ct) {
            const __bf16* bp = sW + ((size_t)(ks * 8 + ct) * 32 + lane) * 16;
            v8bf blo = *(const v8bf*)bp;
            v8bf bhi = *(const v8bf*)(bp + 8);
            v16bf b = combine16(blo, bhi);
            acc[ct] = __builtin_amdgcn_wmma_f32_16x16x32_bf16(
                false, a, false, b, (short)0, acc[ct], false, false);
        }
    }

    // epilogue: C layout — VGPR r: lanes 0-15 M=r, lanes 16-31 M=r+8; N=lane&15
#pragma unroll
    for (int ct = 0; ct < 8; ++ct) {
        int col = ct * 16 + l15;
        float bv = bias[col];
#pragma unroll
        for (int r = 0; r < 8; ++r) {
            int row = rb + r + lhalf * 8;
            float v = acc[ct][r] + bv;
            if (doRelu) {
                v = fmaxf(v, 0.0f);
                outB[(size_t)row * DDIM + col] = (__bf16)v;
            } else {
                outF[(size_t)row * DDIM + col] = v;
            }
        }
    }
}

// ----------------------------------------------------------------- launch
extern "C" void kernel_launch(void* const* d_in, const int* in_sizes, int n_in,
                              void* d_out, int out_size, void* d_ws, size_t ws_size,
                              hipStream_t stream) {
    const float*     x    = (const float*)d_in[0];
    const long long* ei   = (const long long*)d_in[1];   // [2, E] int64
    const float*     W1l  = (const float*)d_in[2];
    const float*     W1r  = (const float*)d_in[3];
    const float*     b1   = (const float*)d_in[4];
    const float*     W2l  = (const float*)d_in[5];
    const float*     W2r  = (const float*)d_in[6];
    const float*     b2   = (const float*)d_in[7];
    float*           out  = (float*)d_out;

    const long long* src = ei;
    const long long* dst = ei + N_EDGES;

    // workspace layout (256B aligned slabs)
    char* ws = (char*)d_ws;
    size_t off = 0;
    auto take = [&](size_t bytes) { char* p = ws + off;
                                    off = (off + bytes + 255) & ~(size_t)255;
                                    return p; };
    __bf16* xb   = (__bf16*)take((size_t)N_NODES * DDIM * 2);
    __bf16* hb   = (__bf16*)take((size_t)N_NODES * DDIM * 2);
    __bf16* mb   = (__bf16*)take((size_t)N_NODES * DDIM * 2);
    float*  agg  = (float*) take((size_t)N_NODES * DDIM * 4);   // cnt follows
    float*  cnt  = (float*) take((size_t)N_NODES * 4);
    __bf16* wf1  = (__bf16*)take((size_t)8 * 8 * 32 * 16 * 2);  // 64 KB
    __bf16* wf2  = (__bf16*)take((size_t)8 * 8 * 32 * 16 * 2);  // 64 KB
    (void)ws_size; (void)n_in; (void)in_sizes; (void)out_size;

    const int nElem  = N_NODES * DDIM;                 // 12.8M
    const int nTiles = N_NODES / 16;                   // 6250

    // 1) x -> bf16
    cvt_f32_to_bf16_kernel<<<nElem / 1024, 256, 0, stream>>>(x, xb, nElem);
    // 2) pack weights into WMMA fragment layouts
    prep_wfrag_kernel<<<8, 256, 0, stream>>>(W1l, W1r, wf1);
    prep_wfrag_kernel<<<8, 256, 0, stream>>>(W2l, W2r, wf2);
    // 3) zero agg + cnt (contiguous)
    {
        int n4 = (nElem + N_NODES) / 4;
        zero_f32_kernel<<<(n4 + 255) / 256, 256, 0, stream>>>(agg, n4);
    }
    // 4) layer-1 aggregation (also builds degree counts)
    aggregate_kernel<<<N_EDGES / 8, 256, 0, stream>>>(xb, src, dst, agg, cnt,
                                                      N_EDGES, 1);
    // 5) mean -> bf16
    finalize_mean_kernel<<<nElem / 1024, 256, 0, stream>>>(agg, cnt, mb);
    // 6) layer-1 GEMM + bias + ReLU -> h (bf16)
    sage_gemm_kernel<<<(nTiles + 7) / 8, 256, 0, stream>>>(
        mb, xb, wf1, b1, nullptr, hb, nTiles, 1);
    // 7) re-zero agg (keep cnt)
    zero_f32_kernel<<<(nElem / 4 + 255) / 256, 256, 0, stream>>>(agg, nElem / 4);
    // 8) layer-2 aggregation from h
    aggregate_kernel<<<N_EDGES / 8, 256, 0, stream>>>(hb, src, dst, agg, cnt,
                                                      N_EDGES, 0);
    // 9) mean -> bf16
    finalize_mean_kernel<<<nElem / 1024, 256, 0, stream>>>(agg, cnt, mb);
    // 10) layer-2 GEMM + bias -> f32 output
    sage_gemm_kernel<<<(nTiles + 7) / 8, 256, 0, stream>>>(
        mb, hb, wf2, b2, out, nullptr, nTiles, 0);
}
